// HRM_5755256176679
// MI455X (gfx1250) — matrix operand order repaired and checked by reading, not compiled, not run on hardware
//
#include <hip/hip_runtime.h>
#include <hip/hip_bf16.h>
#include <math.h>

typedef _Float16 v16h __attribute__((ext_vector_type(16)));
typedef _Float16 v8h  __attribute__((ext_vector_type(8)));
typedef float    v8f  __attribute__((ext_vector_type(8)));

union Frag { v16h v; v8h h[2]; };

__device__ __forceinline__ v8f zero8() {
    v8f z = {0.f,0.f,0.f,0.f,0.f,0.f,0.f,0.f};
    return z;
}

// A-matrix 16x32 f16 fragment (per CDNA5 ISA layout):
// lanes 0-15  hold row m=lane,    K = {k0..k0+7, k0+16..k0+23}
// lanes 16-31 hold row m=lane-16, K = {k0+8..k0+15, k0+24..k0+31}
__device__ __forceinline__ v16h lds_a_frag(const _Float16* base, int m, int feat,
                                           int k0, int lane) {
    int kb = k0 + ((lane & 16) ? 8 : 0);
    const _Float16* p = base + m * feat + kb;
    Frag f;
    f.h[0] = *(const v8h*)(p);
    f.h[1] = *(const v8h*)(p + 16);
    return f.v;
}

// B-matrix 32x16 f16 fragment:
// lanes 0-15  hold col n=lane,    K = {k0..k0+7, k0+16..k0+23}
// lanes 16-31 hold col n=lane-16, K = {k0+8..k0+15, k0+24..k0+31}
// B[k][n] = w[n][k] for inp @ w.T, so each lane reads two runs of 8 contiguous f16 of row n.
__device__ __forceinline__ v16h glb_b_frag(const _Float16* w, int nrow, int K,
                                           int k0, int lane) {
    int kb = k0 + ((lane & 16) ? 8 : 0);
    const _Float16* p = w + (size_t)nrow * K + kb;
    Frag f;
    f.h[0] = *(const v8h*)(p);
    f.h[1] = *(const v8h*)(p + 16);
    return f.v;
}

#define WMMA(a, b, c) \
    __builtin_amdgcn_wmma_f32_16x16x32_f16(false, (a), false, (b), (short)0, (c), false, false)

// One 16(M) x 16(N) tile of a fused GRU cell update.
// ih input is the concatenation [A0 | A1 | A2] (each 256 features wide, LDS f16).
// H = current hidden (256 wide, LDS f16). Writes new hidden tile to Hdst.
__device__ __forceinline__ void gru_tile(
    const _Float16* A0, const _Float16* A1, const _Float16* A2,
    const _Float16* H,
    const _Float16* __restrict__ w_ih, int Kih,
    const _Float16* __restrict__ w_hh,
    const float* __restrict__ b_ih, const float* __restrict__ b_hh,
    _Float16* Hdst, int mbase, int noff, int lane)
{
    int nl   = lane & 15;
    int am   = mbase + nl;
    int nrow = noff + nl;
    v8f ir = zero8(), iz = zero8(), in_ = zero8();
    v8f hr = zero8(), hz = zero8(), hn = zero8();

    int nkt = Kih / 32;
    for (int kt = 0; kt < nkt; ++kt) {
        const _Float16* src = (kt < 8) ? A0 : (kt < 16) ? A1 : A2;
        v16h a  = lds_a_frag(src, am, 256, (kt & 7) * 32, lane);
        v16h b0 = glb_b_frag(w_ih, nrow,       Kih, kt * 32, lane);
        v16h b1 = glb_b_frag(w_ih, nrow + 256, Kih, kt * 32, lane);
        v16h b2 = glb_b_frag(w_ih, nrow + 512, Kih, kt * 32, lane);
        ir  = WMMA(a, b0, ir);
        iz  = WMMA(a, b1, iz);
        in_ = WMMA(a, b2, in_);
    }
    for (int kt = 0; kt < 8; ++kt) {
        v16h a  = lds_a_frag(H, am, 256, kt * 32, lane);
        v16h b0 = glb_b_frag(w_hh, nrow,       256, kt * 32, lane);
        v16h b1 = glb_b_frag(w_hh, nrow + 256, 256, kt * 32, lane);
        v16h b2 = glb_b_frag(w_hh, nrow + 512, 256, kt * 32, lane);
        hr = WMMA(a, b0, hr);
        hz = WMMA(a, b1, hz);
        hn = WMMA(a, b2, hn);
    }

    float bir = b_ih[nrow], biz = b_ih[nrow + 256], bin = b_ih[nrow + 512];
    float bhr = b_hh[nrow], bhz = b_hh[nrow + 256], bhn = b_hh[nrow + 512];
    int r0 = mbase + ((lane & 16) ? 8 : 0);   // C/D layout: VGPR v -> row r0+v
#pragma unroll
    for (int v = 0; v < 8; ++v) {
        int row   = r0 + v;
        float rg  = (ir[v] + bir) + (hr[v] + bhr);
        float zg  = (iz[v] + biz) + (hz[v] + bhz);
        float r   = 1.f / (1.f + __expf(-rg));
        float z   = 1.f / (1.f + __expf(-zg));
        float ng  = tanhf((in_[v] + bin) + r * (hn[v] + bhn));
        float hold = (float)H[row * 256 + nrow];
        Hdst[row * 256 + nrow] = (_Float16)((1.f - z) * ng + z * hold);
    }
}

__global__ __launch_bounds__(256, 1) void hrm_fused_kernel(
    const float* __restrict__ x,
    const float* __restrict__ proj_b,
    const float* __restrict__ lbih, const float* __restrict__ lbhh,
    const float* __restrict__ hbih, const float* __restrict__ hbhh,
    const float* __restrict__ ln_g, const float* __restrict__ ln_b,
    const float* __restrict__ out_w, const float* __restrict__ out_b,
    const _Float16* __restrict__ wp,    // 256x512 f16
    const _Float16* __restrict__ wlih,  // 768x768 f16
    const _Float16* __restrict__ wlhh,  // 768x256 f16
    const _Float16* __restrict__ whih,  // 768x512 f16
    const _Float16* __restrict__ whhh,  // 768x256 f16
    float* __restrict__ out)
{
    // 160 KB LDS: xe (32K) | zh0 | zh1 | zl0 | zl1 (32K each)
    __shared__ __align__(16) _Float16 lds[81920];
    // NOTE: compute LDS-derived pointers on the fly (no pointer arrays —
    // constant-initialized LDS pointer arrays fail to link on gfx1250).
#define XE_BUF   (lds)
#define ZH_BUF(i) (lds + 16384 + (i) * 16384)
#define ZL_BUF(i) (lds + 49152 + (i) * 16384)
#define XST_BUF  (lds + 16384)   // x staging (64x512 f16) overlays ZH0+ZH1

    int tid   = threadIdx.x;
    int lane  = tid & 31;
    int wave  = tid >> 5;            // 8 wave32s
    int mbase = (wave >> 1) * 16;    // 4 M-subtiles of 16 rows
    int nhalf = wave & 1;            // 2 N-halves of 128 cols
    size_t blk = blockIdx.x;
    const float* xg = x + blk * (64ull * 512);

    // ---- stage x tile (64x512 f32 -> f16 in LDS) ----
    for (int i = tid * 4; i < 64 * 512; i += 256 * 4) {
        float4 v = *(const float4*)(xg + i);
        XST_BUF[i + 0] = (_Float16)v.x; XST_BUF[i + 1] = (_Float16)v.y;
        XST_BUF[i + 2] = (_Float16)v.z; XST_BUF[i + 3] = (_Float16)v.w;
    }
    __syncthreads();

    // ---- x_embed = gelu(x @ proj_w.T + proj_b) ----
    for (int t = 0; t < 8; ++t) {
        int noff = nhalf * 128 + t * 16;
        int nl   = lane & 15;
        int am   = mbase + nl;
        int nrow = noff + nl;
        v8f acc = zero8();
        for (int kt = 0; kt < 16; ++kt) {
            v16h a = lds_a_frag(XST_BUF, am, 512, kt * 32, lane);
            v16h b = glb_b_frag(wp, nrow, 512, kt * 32, lane);
            acc = WMMA(a, b, acc);
        }
        float bias = proj_b[nrow];
        int r0 = mbase + ((lane & 16) ? 8 : 0);
#pragma unroll
        for (int v = 0; v < 8; ++v) {
            float y = acc[v] + bias;
            float g = 0.5f * y * (1.f + erff(y * 0.70710678118f));  // exact GELU
            XE_BUF[(r0 + v) * 256 + nrow] = (_Float16)g;
        }
    }
    __syncthreads();

    // ---- zero initial states z_h, z_l ----
    for (int i = tid; i < 8192; i += 256) {
        ((unsigned*)ZH_BUF(0))[i] = 0u;
        ((unsigned*)ZL_BUF(0))[i] = 0u;
    }
    __syncthreads();

    // ---- 12 low-GRU steps; high-GRU after steps 4, 8, 12 ----
    int cl = 0, ch = 0;
    for (int s = 1; s <= 12; ++s) {
        for (int t = 0; t < 8; ++t) {
            int noff = nhalf * 128 + t * 16;
            // l_input = [x_embed | z_h | z_l], hidden = z_l, K_ih = 768
            gru_tile(XE_BUF, ZH_BUF(ch), ZL_BUF(cl), ZL_BUF(cl), wlih, 768, wlhh,
                     lbih, lbhh, ZL_BUF(cl ^ 1), mbase, noff, lane);
        }
        __syncthreads();
        cl ^= 1;
        if ((s & 3) == 0) {
            for (int t = 0; t < 8; ++t) {
                int noff = nhalf * 128 + t * 16;
                // h_input = [z_l_new | z_h], hidden = z_h, K_ih = 512
                gru_tile(ZL_BUF(cl), ZH_BUF(ch), (const _Float16*)0, ZH_BUF(ch),
                         whih, 512, whhh, hbih, hbhh, ZH_BUF(ch ^ 1),
                         mbase, noff, lane);
            }
            __syncthreads();
            ch ^= 1;
        }
    }

    // ---- write z_h_new, z_l_new (f16 -> f32) ----
    size_t rb = blk * (64ull * 256);
    const size_t ZLOFF = 65536ull * 256;
    const size_t LGOFF = 2ull * ZLOFF;
    for (int i = tid; i < 64 * 256; i += 256) {
        out[rb + i]         = (float)ZH_BUF(ch)[i];
        out[ZLOFF + rb + i] = (float)ZL_BUF(cl)[i];
    }

    // ---- LayerNorm + Linear(256 -> 2) head; one wave per 8 rows ----
    for (int rr = 0; rr < 8; ++rr) {
        int row = wave * 8 + rr;
        const _Float16* zr = ZH_BUF(ch) + row * 256;
        float vals[8];
        float s = 0.f;
#pragma unroll
        for (int j = 0; j < 8; ++j) { vals[j] = (float)zr[lane + j * 32]; s += vals[j]; }
#pragma unroll
        for (int off = 16; off > 0; off >>= 1) s += __shfl_xor(s, off, 32);
        float mu = s * (1.f / 256.f);
        float vsum = 0.f;
#pragma unroll
        for (int j = 0; j < 8; ++j) { float d = vals[j] - mu; vsum += d * d; }
#pragma unroll
        for (int off = 16; off > 0; off >>= 1) vsum += __shfl_xor(vsum, off, 32);
        float rstd = rsqrtf(vsum * (1.f / 256.f) + 1e-5f);
        float l0 = 0.f, l1 = 0.f;
#pragma unroll
        for (int j = 0; j < 8; ++j) {
            int c = lane + j * 32;
            float nv = (vals[j] - mu) * rstd * ln_g[c] + ln_b[c];
            l0 += nv * out_w[c];
            l1 += nv * out_w[256 + c];
        }
#pragma unroll
        for (int off = 16; off > 0; off >>= 1) {
            l0 += __shfl_xor(l0, off, 32);
            l1 += __shfl_xor(l1, off, 32);
        }
        if (lane == 0) {
            size_t r = blk * 64 + row;
            out[LGOFF + r * 2 + 0] = l0 + out_b[0];
            out[LGOFF + r * 2 + 1] = l1 + out_b[1];
        }
    }
#undef XE_BUF
#undef ZH_BUF
#undef ZL_BUF
#undef XST_BUF
}

__global__ void cvt_f32_f16_kernel(const float* __restrict__ src,
                                   _Float16* __restrict__ dst, int n) {
    int i = blockIdx.x * blockDim.x + threadIdx.x;
    if (i < n) dst[i] = (_Float16)src[i];
}

extern "C" void kernel_launch(void* const* d_in, const int* in_sizes, int n_in,
                              void* d_out, int out_size, void* d_ws, size_t ws_size,
                              hipStream_t stream) {
    const float* x      = (const float*)d_in[0];
    const float* proj_w = (const float*)d_in[1];
    const float* proj_b = (const float*)d_in[2];
    const float* lwih   = (const float*)d_in[3];
    const float* lwhh   = (const float*)d_in[4];
    const float* lbih   = (const float*)d_in[5];
    const float* lbhh   = (const float*)d_in[6];
    const float* hwih   = (const float*)d_in[7];
    const float* hwhh   = (const float*)d_in[8];
    const float* hbih   = (const float*)d_in[9];
    const float* hbhh   = (const float*)d_in[10];
    const float* ln_g   = (const float*)d_in[11];
    const float* ln_b   = (const float*)d_in[12];
    const float* out_w  = (const float*)d_in[13];
    const float* out_b  = (const float*)d_in[14];

    // f16 weight cache in workspace (~3 MB, L2-resident at run time)
    _Float16* ws   = (_Float16*)d_ws;
    _Float16* wp   = ws;            // 256*512  = 131072
    _Float16* wlih = ws + 131072;   // 768*768  = 589824
    _Float16* wlhh = ws + 720896;   // 768*256  = 196608
    _Float16* whih = ws + 917504;   // 768*512  = 393216
    _Float16* whhh = ws + 1310720;  // 768*256  = 196608

    cvt_f32_f16_kernel<<<(131072 + 255) / 256, 256, 0, stream>>>(proj_w, wp, 131072);
    cvt_f32_f16_kernel<<<(589824 + 255) / 256, 256, 0, stream>>>(lwih, wlih, 589824);
    cvt_f32_f16_kernel<<<(196608 + 255) / 256, 256, 0, stream>>>(lwhh, wlhh, 196608);
    cvt_f32_f16_kernel<<<(393216 + 255) / 256, 256, 0, stream>>>(hwih, whih, 393216);
    cvt_f32_f16_kernel<<<(196608 + 255) / 256, 256, 0, stream>>>(hwhh, whhh, 196608);

    // 65536 rows / 64 rows per workgroup = 1024 workgroups, 256 threads (8 wave32)
    hrm_fused_kernel<<<1024, 256, 0, stream>>>(
        x, proj_b, lbih, lbhh, hbih, hbhh, ln_g, ln_b, out_w, out_b,
        wp, wlih, wlhh, whih, whhh, (float*)d_out);
}
